// BaselineSelfAttention_14843406975367
// MI455X (gfx1250) — matrix operand order repaired and checked by reading, not compiled, not run on hardware
//
#include <hip/hip_runtime.h>
#include <hip/hip_bf16.h>

// ---------------------------------------------------------------------------
// CDNA5 (gfx1250) multi-query attention, bf16 WMMA (v_wmma_f32_16x16x32_bf16)
// ---------------------------------------------------------------------------
typedef __attribute__((ext_vector_type(16))) __bf16 v16bf;
typedef __attribute__((ext_vector_type(8)))  __bf16 v8bf;
typedef __attribute__((ext_vector_type(4)))  __bf16 v4bf;
typedef __attribute__((ext_vector_type(8)))  float  v8f;

union V16U { v16bf v; v8bf h[2]; };

#define WMMA_BF16(a, b, c) \
  __builtin_amdgcn_wmma_f32_16x16x32_bf16(false, (a), false, (b), (short)0, (c), false, false)

// ---------------------------------------------------------------------------
// fp32 -> bf16 conversion, 4 elems/thread (all sizes are multiples of 4)
// ---------------------------------------------------------------------------
__global__ void cvt_f32_bf16(const float* __restrict__ s, __bf16* __restrict__ d, int n4) {
  int i = blockIdx.x * blockDim.x + threadIdx.x;
  int stride = gridDim.x * blockDim.x;
  for (; i < n4; i += stride) {
    const float4 f = ((const float4*)s)[i];
    v4bf o; o[0] = (__bf16)f.x; o[1] = (__bf16)f.y; o[2] = (__bf16)f.z; o[3] = (__bf16)f.w;
    ((v4bf*)d)[i] = o;
  }
}

// ---------------------------------------------------------------------------
// GEMM: out = (A[M,K] * W[N,K]^T + bias[N]) * osc
//   mode 0: bf16 out, row-major [M,N]
//   mode 1: bf16 out, scatter to q_perm[B,H=16,S,Dh=64], h=n%16, dh=n/16
//   mode 2: f32  out, row-major [M,N]
// 128 threads = 4 waves; each wave computes a 32x64 tile (2 A-frags x 4
// B-frags -> 8 WMMAs per 32-wide K chunk). Block covers 128x64.
// ---------------------------------------------------------------------------
__global__ void __launch_bounds__(128)
gemm_bias(const __bf16* __restrict__ A, const __bf16* __restrict__ W,
          const float* __restrict__ bias, void* __restrict__ out,
          int M, int N, int K, int mode, int S, float osc)
{
  const int tid  = threadIdx.x;
  const int wave = tid >> 5;
  const int lane = tid & 31;
  const int g    = lane >> 4;     // half-wave group
  const int ln   = lane & 15;

  const int nb = blockIdx.x * 64;            // 64-wide N tile per block
  const int mb = blockIdx.y * 128 + wave * 32;  // 32-high M tile per wave

  const __bf16* arow0 = A + (size_t)(mb + ln) * K;        // A-frag rows
  const __bf16* arow1 = A + (size_t)(mb + 16 + ln) * K;
  const __bf16* wrow0 = W + (size_t)(nb + ln) * K;        // B-frag cols
  const __bf16* wrow1 = W + (size_t)(nb + 16 + ln) * K;
  const __bf16* wrow2 = W + (size_t)(nb + 32 + ln) * K;
  const __bf16* wrow3 = W + (size_t)(nb + 48 + ln) * K;

  v8f acc[2][4] = {{{}, {}, {}, {}}, {{}, {}, {}, {}}};

  for (int kb = 0; kb < K; kb += 32) {
    __builtin_prefetch(arow0 + kb + 256, 0, 3);   // global_prefetch_b8
    __builtin_prefetch(arow1 + kb + 256, 0, 3);

    V16U a0, a1;
    a0.h[0] = *(const v8bf*)(arow0 + kb + 16 * g);      // k = 16g + e
    a0.h[1] = *(const v8bf*)(arow0 + kb + 16 * g + 8);  // k = 16g + 8 + e
    a1.h[0] = *(const v8bf*)(arow1 + kb + 16 * g);
    a1.h[1] = *(const v8bf*)(arow1 + kb + 16 * g + 8);

    const v16bf b0 = *(const v16bf*)(wrow0 + kb + 16 * g);  // k = 16g + e
    const v16bf b1 = *(const v16bf*)(wrow1 + kb + 16 * g);
    const v16bf b2 = *(const v16bf*)(wrow2 + kb + 16 * g);
    const v16bf b3 = *(const v16bf*)(wrow3 + kb + 16 * g);

    acc[0][0] = WMMA_BF16(a0.v, b0, acc[0][0]);
    acc[0][1] = WMMA_BF16(a0.v, b1, acc[0][1]);
    acc[0][2] = WMMA_BF16(a0.v, b2, acc[0][2]);
    acc[0][3] = WMMA_BF16(a0.v, b3, acc[0][3]);
    acc[1][0] = WMMA_BF16(a1.v, b0, acc[1][0]);
    acc[1][1] = WMMA_BF16(a1.v, b1, acc[1][1]);
    acc[1][2] = WMMA_BF16(a1.v, b2, acc[1][2]);
    acc[1][3] = WMMA_BF16(a1.v, b3, acc[1][3]);
  }

#pragma unroll
  for (int nt = 0; nt < 4; ++nt) {
    const int n = nb + nt * 16 + ln;
    const float bv = bias[n];
#pragma unroll
    for (int mi = 0; mi < 2; ++mi) {
#pragma unroll
      for (int r = 0; r < 8; ++r) {
        const int m = mb + mi * 16 + r + 8 * g;          // C layout: m = r + 8g
        const float val = (acc[mi][nt][r] + bv) * osc;
        if (mode == 0) {
          ((__bf16*)out)[(size_t)m * N + n] = (__bf16)val;
        } else if (mode == 1) {
          const int hh = n & 15, dh = n >> 4;            // reference reshape (Dh,H)
          const int s  = m % S,  bb = m / S;
          ((__bf16*)out)[(((size_t)(bb * 16 + hh)) * S + s) * 64 + dh] = (__bf16)val;
        } else {
          ((float*)out)[(size_t)m * N + n] = val;
        }
      }
    }
  }
}

// ---------------------------------------------------------------------------
// Fused MQA flash attention.
//   qp  : [B, H, S, 64] bf16 (pre-scaled by 1/sqrt(Dh) in projection epilogue)
//   kbuf: [B, S, 64]    bf16 (single KV head)
//   vbuf: [B, S, 64]    bf16
//   mrg : [B*S, H*64]   bf16, merged-head order (h*64 + dh)
// 256 threads = 8 waves; wave w handles query rows [q0, q0+16) of one (b,h).
// Block streams shared K/V 32-key tiles through LDS.
// ---------------------------------------------------------------------------
__global__ void __launch_bounds__(256)
mqa_attention(const __bf16* __restrict__ qp, const __bf16* __restrict__ kbuf,
              const __bf16* __restrict__ vbuf, __bf16* __restrict__ mrg,
              int B, int Hn, int S)
{
  __shared__ __align__(32) __bf16 Kt[32 * 64];        // [key][dh]
  __shared__ __align__(32) __bf16 Vt[64 * 32];        // [dh][key] (transposed)
  __shared__ __align__(32) __bf16 Pst[8][16 * 32];    // per-wave P staging

  const int tid  = threadIdx.x;
  const int wave = tid >> 5;
  const int lane = tid & 31;
  const int g    = lane >> 4;
  const int ln   = lane & 15;

  const int qblocks = S / 128;
  const int qblk = blockIdx.x % qblocks;
  const int bh   = blockIdx.x / qblocks;
  const int h    = bh % Hn;
  const int b    = bh / Hn;
  const int q0   = qblk * 128 + wave * 16;

  // Q fragments: 16x64 -> two 16x32 A-frags
  const __bf16* qrow = qp + (((size_t)(b * Hn + h)) * S + q0 + ln) * 64;
  V16U qa[2];
  qa[0].h[0] = *(const v8bf*)(qrow + 0  + 16 * g);
  qa[0].h[1] = *(const v8bf*)(qrow + 8  + 16 * g);
  qa[1].h[0] = *(const v8bf*)(qrow + 32 + 16 * g);
  qa[1].h[1] = *(const v8bf*)(qrow + 40 + 16 * g);

  float mr[8], lr[8];
  v8f o[4] = {{}, {}, {}, {}};
#pragma unroll
  for (int r = 0; r < 8; ++r) { mr[r] = -1e30f; lr[r] = 0.f; }

  // cooperative staging mapping: 256 threads cover 32x64 tile, 8 elems each
  const int srow = tid >> 3;
  const int scol = (tid & 7) * 8;

  for (int kb = 0; kb < S; kb += 32) {
    __syncthreads();
    {
      const size_t base = ((size_t)b * S + kb + srow) * 64 + scol;
      const v8bf kv = *(const v8bf*)(kbuf + base);
      const v8bf vv = *(const v8bf*)(vbuf + base);
      *(v8bf*)(Kt + srow * 64 + scol) = kv;
#pragma unroll
      for (int j = 0; j < 8; ++j) Vt[(scol + j) * 32 + srow] = vv[j];
    }
    __syncthreads();

    // ---- scores: S16x32 = Qs(16x64) . K^T, as two 16x16 C tiles ----
    v8f s0 = {}, s1 = {};
#pragma unroll
    for (int kk = 0; kk < 2; ++kk) {
      const v16bf b0 = *(const v16bf*)(Kt + ln * 64        + kk * 32 + 16 * g);
      const v16bf b1 = *(const v16bf*)(Kt + (16 + ln) * 64 + kk * 32 + 16 * g);
      s0 = WMMA_BF16(qa[kk].v, b0, s0);
      s1 = WMMA_BF16(qa[kk].v, b1, s1);
    }

    // ---- online softmax (per row m = r + 8g, reduce across 16 lanes) ----
#pragma unroll
    for (int r = 0; r < 8; ++r) {
      const float a0 = s0[r], a1 = s1[r];     // scale already folded into Q
      float mx = fmaxf(a0, a1);
      mx = fmaxf(mx, __shfl_xor(mx, 1, 16));
      mx = fmaxf(mx, __shfl_xor(mx, 2, 16));
      mx = fmaxf(mx, __shfl_xor(mx, 4, 16));
      mx = fmaxf(mx, __shfl_xor(mx, 8, 16));
      const float nm = fmaxf(mr[r], mx);
      const float p0 = __expf(a0 - nm);
      const float p1 = __expf(a1 - nm);
      float rs = p0 + p1;
      rs += __shfl_xor(rs, 1, 16);
      rs += __shfl_xor(rs, 2, 16);
      rs += __shfl_xor(rs, 4, 16);
      rs += __shfl_xor(rs, 8, 16);
      const float alpha = __expf(mr[r] - nm);
      lr[r] = lr[r] * alpha + rs;
      mr[r] = nm;
#pragma unroll
      for (int t = 0; t < 4; ++t) o[t][r] *= alpha;
      const int m = r + 8 * g;
      Pst[wave][m * 32 + ln]      = (__bf16)p0;        // keys kb..kb+15
      Pst[wave][m * 32 + 16 + ln] = (__bf16)p1;        // keys kb+16..kb+31
    }

    // ---- O += P(16x32) . V(32x64) ----
    V16U pa;
    pa.h[0] = *(const v8bf*)(&Pst[wave][ln * 32 + 16 * g]);
    pa.h[1] = *(const v8bf*)(&Pst[wave][ln * 32 + 16 * g + 8]);
#pragma unroll
    for (int t = 0; t < 4; ++t) {
      const v16bf vb = *(const v16bf*)(Vt + (t * 16 + ln) * 32 + 16 * g);
      o[t] = WMMA_BF16(pa.v, vb, o[t]);
    }
  }

  // ---- epilogue: merged[b*S + q0 + m][h*64 + t*16 + ln] = O / l ----
#pragma unroll
  for (int r = 0; r < 8; ++r) {
    const int m = r + 8 * g;
    const float inv = 1.0f / lr[r];
    const size_t row = ((size_t)b * S + q0 + m) * (size_t)(Hn * 64);
#pragma unroll
    for (int t = 0; t < 4; ++t)
      mrg[row + h * 64 + t * 16 + ln] = (__bf16)(o[t][r] * inv);
  }
}

// ---------------------------------------------------------------------------
// Host driver
// ---------------------------------------------------------------------------
extern "C" void kernel_launch(void* const* d_in, const int* in_sizes, int n_in,
                              void* d_out, int out_size, void* d_ws, size_t ws_size,
                              hipStream_t stream) {
  (void)in_sizes; (void)n_in; (void)out_size; (void)ws_size;
  const float* x  = (const float*)d_in[0];
  const float* wq = (const float*)d_in[1];
  const float* bq = (const float*)d_in[2];
  const float* wk = (const float*)d_in[3];
  const float* bk = (const float*)d_in[4];
  const float* wv = (const float*)d_in[5];
  const float* bv = (const float*)d_in[6];
  const float* wo = (const float*)d_in[7];
  const float* bo = (const float*)d_in[8];

  const int B = 2, S = 2048, D = 1024, Hn = 16, Dh = 64;
  const int M = B * S;  // 4096

  size_t off = 0;
  auto alloc = [&](size_t elems) -> __bf16* {
    __bf16* p = (__bf16*)((char*)d_ws + off);
    off += elems * sizeof(__bf16);
    off = (off + 255) & ~(size_t)255;
    return p;
  };
  __bf16* x_bf  = alloc((size_t)M * D);           // inputs bf16
  __bf16* wq_bf = alloc((size_t)D * D);
  __bf16* wk_bf = alloc((size_t)Dh * D);
  __bf16* wv_bf = alloc((size_t)Dh * D);
  __bf16* wo_bf = alloc((size_t)D * D);
  __bf16* q_p   = alloc((size_t)B * Hn * S * Dh); // Q permuted [B,H,S,Dh]
  __bf16* k_bf  = alloc((size_t)B * S * Dh);
  __bf16* v_bf  = alloc((size_t)B * S * Dh);
  __bf16* mrg   = alloc((size_t)M * D);           // merged heads

  auto cvt = [&](const float* s, __bf16* d, int n) {
    const int n4 = n / 4;
    int blocks = (n4 + 255) / 256;
    if (blocks > 2048) blocks = 2048;
    cvt_f32_bf16<<<blocks, 256, 0, stream>>>(s, d, n4);
  };
  cvt(x,  x_bf,  M * D);
  cvt(wq, wq_bf, D * D);
  cvt(wk, wk_bf, Dh * D);
  cvt(wv, wv_bf, Dh * D);
  cvt(wo, wo_bf, D * D);

  const dim3 blk(128);
  const float qscale = 0.125f;  // 1/sqrt(Dh), folded into Q so softmax sees scaled scores
  // Q projection: scatter into [B,H,S,Dh] honoring (Dh,H) column split, pre-scaled
  gemm_bias<<<dim3(D / 64, M / 128), blk, 0, stream>>>(x_bf, wq_bf, bq, q_p, M, D,  D, 1, S, qscale);
  // K / V projections (single KV head)
  gemm_bias<<<dim3(1,      M / 128), blk, 0, stream>>>(x_bf, wk_bf, bk, k_bf, M, Dh, D, 0, S, 1.0f);
  gemm_bias<<<dim3(1,      M / 128), blk, 0, stream>>>(x_bf, wv_bf, bv, v_bf, M, Dh, D, 0, S, 1.0f);

  // Fused flash MQA attention -> merged heads
  mqa_attention<<<B * Hn * (S / 128), 256, 0, stream>>>(q_p, k_bf, v_bf, mrg, B, Hn, S);

  // Output projection (fp32 result)
  gemm_bias<<<dim3(D / 64, M / 128), blk, 0, stream>>>(mrg, wo_bf, bo, d_out, M, D, D, 2, S, 1.0f);
}
